// TimeShiftedCrossModalAttention_90288802497059
// MI455X (gfx1250) — matrix operand order repaired and checked by reading, not compiled, not run on hardware
//
#include <hip/hip_runtime.h>
#include <math.h>

// Problem constants (from reference)
static constexpr int BATCH    = 2;
static constexpr int T        = 2048;
static constexpr int D        = 512;
static constexpr int HEADS    = 8;
static constexpr int DIM_HEAD = 64;
static constexpr int MAX_LAG  = 5;
static constexpr int INNER    = HEADS * DIM_HEAD;   // 512
static constexpr int MROWS    = BATCH * T;          // 4096

typedef __attribute__((ext_vector_type(16))) _Float16 v16h;
typedef __attribute__((ext_vector_type(8)))  _Float16 v8h;
typedef __attribute__((ext_vector_type(4)))  _Float16 v4h;
typedef __attribute__((ext_vector_type(8)))  float    v8f;
typedef __attribute__((ext_vector_type(4)))  float    v4f;

__device__ __forceinline__ int lane_id() { return (int)(threadIdx.x & 31); }

// ---------------------------------------------------------------------------
// WMMA fragment loaders (wave32, v_wmma_f32_16x16x32_f16)
// A (16xK rows=M): lane L holds row L%16; 16 contiguous halves at k=16*(L/16).
// B (Kx16 cols=N): lane L holds col L%16; 16 contiguous halves at k=16*(L/16).
//   Loaded from BT[N][K] row-major (i.e. B column-major).
// ---------------------------------------------------------------------------
__device__ __forceinline__ v16h load_a_f16(const _Float16* p, int ld) {
  const int l  = lane_id();
  const int r  = l & 15;
  const int kh = (l >> 4) << 4;                 // 0 or 16
  union { v16h v; v8h h[2]; } u;
  u.h[0] = *(const v8h*)(p + (long)r * ld + kh);
  u.h[1] = *(const v8h*)(p + (long)r * ld + kh + 8);
  return u.v;
}

__device__ __forceinline__ v16h load_b_f16(const _Float16* p, int ld) {
  const int l  = lane_id();
  const int r  = l & 15;
  const int kh = (l >> 4) << 4;                 // 0 or 16
  return *(const v16h*)(p + (long)r * ld + kh);
}

__device__ __forceinline__ v8f wmma_f16(v16h a, v16h b, v8f c) {
  return __builtin_amdgcn_wmma_f32_16x16x32_f16(false, a, false, b, (short)0, c,
                                                false, false);
}

// ---------------------------------------------------------------------------
// Kernel 0: x f32 -> f16, coalesced (4 elems/thread)
// ---------------------------------------------------------------------------
__global__ void xcvt_kernel(const float* __restrict__ X, _Float16* __restrict__ Xh,
                            int n4) {
  int idx = blockIdx.x * blockDim.x + threadIdx.x;
  if (idx >= n4) return;
  const long base = (long)idx * 4;
  v4f xv = *(const v4f*)(X + base);
  v4h o;
#pragma unroll
  for (int i = 0; i < 4; ++i) o[i] = (_Float16)xv[i];
  *(v4h*)(Xh + base) = o;
}

// ---------------------------------------------------------------------------
// Kernel 1: weight convert + transpose  WT[n*K + k] = (f16) W[k*N + n]
// ---------------------------------------------------------------------------
__global__ void wtrans_kernel(const float* __restrict__ W, _Float16* __restrict__ WT,
                              int K, int N) {
  int idx = blockIdx.x * blockDim.x + threadIdx.x;
  if (idx >= K * N) return;
  int n = idx / K;
  int k = idx - n * K;
  WT[idx] = (_Float16)W[(long)k * N + n];
}

// ---------------------------------------------------------------------------
// Kernel 2: QKV projection GEMM (f16 A from Xh).  M=4096, N=512, K=512.
// Block = 4 waves; each wave computes a 32(M) x 64(N) tile, double-buffered.
// vmode=0 : store f16 per-head row-major  [b,h,t,dh]
// vmode=1 : store f16 transposed          [b,h,dh,t]   (for V)
// ---------------------------------------------------------------------------
__global__ void __launch_bounds__(128, 1)
proj_qkv_kernel(const _Float16* __restrict__ Xh,
                const _Float16* __restrict__ BT,
                _Float16* __restrict__ out, int vmode) {
  const int w  = threadIdx.x >> 5;
  const int l  = threadIdx.x & 31;
  const int i0 = blockIdx.x * 128 + w * 32;
  const int n0 = blockIdx.y * 64;
  v8f acc0[4] = {}, acc1[4] = {};

  v16h a0 = load_a_f16(Xh + (long)i0 * D, D);
  v16h a1 = load_a_f16(Xh + (long)(i0 + 16) * D, D);
  v16h b0 = load_b_f16(BT + (long)(n0 +  0) * D, D);
  v16h b1 = load_b_f16(BT + (long)(n0 + 16) * D, D);
  v16h b2 = load_b_f16(BT + (long)(n0 + 32) * D, D);
  v16h b3 = load_b_f16(BT + (long)(n0 + 48) * D, D);

  for (int k0 = 32; k0 <= D - 32; k0 += 32) {
    v16h na0 = load_a_f16(Xh + (long)i0 * D + k0, D);
    v16h na1 = load_a_f16(Xh + (long)(i0 + 16) * D + k0, D);
    v16h nb0 = load_b_f16(BT + (long)(n0 +  0) * D + k0, D);
    v16h nb1 = load_b_f16(BT + (long)(n0 + 16) * D + k0, D);
    v16h nb2 = load_b_f16(BT + (long)(n0 + 32) * D + k0, D);
    v16h nb3 = load_b_f16(BT + (long)(n0 + 48) * D + k0, D);
    acc0[0] = wmma_f16(a0, b0, acc0[0]);
    acc0[1] = wmma_f16(a0, b1, acc0[1]);
    acc0[2] = wmma_f16(a0, b2, acc0[2]);
    acc0[3] = wmma_f16(a0, b3, acc0[3]);
    acc1[0] = wmma_f16(a1, b0, acc1[0]);
    acc1[1] = wmma_f16(a1, b1, acc1[1]);
    acc1[2] = wmma_f16(a1, b2, acc1[2]);
    acc1[3] = wmma_f16(a1, b3, acc1[3]);
    a0 = na0; a1 = na1; b0 = nb0; b1 = nb1; b2 = nb2; b3 = nb3;
  }
  acc0[0] = wmma_f16(a0, b0, acc0[0]);
  acc0[1] = wmma_f16(a0, b1, acc0[1]);
  acc0[2] = wmma_f16(a0, b2, acc0[2]);
  acc0[3] = wmma_f16(a0, b3, acc0[3]);
  acc1[0] = wmma_f16(a1, b0, acc1[0]);
  acc1[1] = wmma_f16(a1, b1, acc1[1]);
  acc1[2] = wmma_f16(a1, b2, acc1[2]);
  acc1[3] = wmma_f16(a1, b3, acc1[3]);

  const int hi = (l >> 4) * 8;
#pragma unroll
  for (int half = 0; half < 2; ++half) {
    const v8f* acc = half ? acc1 : acc0;
    const int ibase = i0 + half * 16;
#pragma unroll
    for (int n = 0; n < 4; ++n) {
      const int c  = n0 + n * 16 + (l & 15);
      const int h  = c >> 6;
      const int dc = c & 63;
#pragma unroll
      for (int v = 0; v < 8; ++v) {
        const int r = ibase + v + hi;
        const int b = r >> 11;          // / T
        const int t = r & (T - 1);      // % T
        long idx;
        if (vmode)
          idx = ((long)((b * HEADS + h) * DIM_HEAD + dc)) * T + t;
        else
          idx = ((long)((b * HEADS + h) * T + t)) * DIM_HEAD + dc;
        out[idx] = (_Float16)acc[n][v];
      }
    }
  }
}

// ---------------------------------------------------------------------------
// Kernel 3: flash attention with causal mask + lag bias.
// Grid: (T/64, B*H); block = 128 threads = 4 waves; wave -> 16 query rows.
// Far chunks (j0 <= i0-36): no mask, constant lag bias lp[5].
// Near-diagonal chunks: full mask + lag-select epilogue.
// ---------------------------------------------------------------------------
__device__ __forceinline__ float lag_sel(const float lp[MAX_LAG + 1], int d) {
  float r = lp[MAX_LAG];
  r = (d == 0) ? lp[0] : r;
  r = (d == 1) ? lp[1] : r;
  r = (d == 2) ? lp[2] : r;
  r = (d == 3) ? lp[3] : r;
  r = (d == 4) ? lp[4] : r;
  return r;
}

__global__ void __launch_bounds__(128, 1)
flash_kernel(const _Float16* __restrict__ Qh,
             const _Float16* __restrict__ Kh,
             const _Float16* __restrict__ Vt,
             const float* __restrict__ lagw,
             _Float16* __restrict__ Ah) {
  __shared__ _Float16 lds[4][16][40];           // padded P staging per wave
  const int w   = threadIdx.x >> 5;
  const int l   = threadIdx.x & 31;
  const int bh  = blockIdx.y;
  const int b   = bh / HEADS;
  const int h   = bh - b * HEADS;
  const int i0  = blockIdx.x * 64 + w * 16;
  const float scale = 0.125f;                   // dh^-0.5

  // per-head lag softmax (6 values)
  float lp[MAX_LAG + 1];
  {
    float mx = -1e30f;
#pragma unroll
    for (int i = 0; i <= MAX_LAG; ++i) {
      lp[i] = lagw[h * (MAX_LAG + 1) + i];
      mx = fmaxf(mx, lp[i]);
    }
    float s = 0.f;
#pragma unroll
    for (int i = 0; i <= MAX_LAG; ++i) { lp[i] = __expf(lp[i] - mx); s += lp[i]; }
    const float inv = 1.f / s;
#pragma unroll
    for (int i = 0; i <= MAX_LAG; ++i) lp[i] *= inv;
  }
  const float lp5 = lp[MAX_LAG];

  const _Float16* Qb = Qh + (long)bh * T * DIM_HEAD;
  const _Float16* Kb = Kh + (long)bh * T * DIM_HEAD;
  const _Float16* Vb = Vt + (long)bh * DIM_HEAD * T;

  const v16h qa0 = load_a_f16(Qb + (long)i0 * DIM_HEAD, DIM_HEAD);
  const v16h qa1 = load_a_f16(Qb + (long)i0 * DIM_HEAD + 32, DIM_HEAD);

  float m[8], lsum[8];
#pragma unroll
  for (int v = 0; v < 8; ++v) { m[v] = -1e30f; lsum[v] = 0.f; }
  v8f o[4] = {};

  const int hi   = (l >> 4) * 8;
  const int col  = l & 15;
  const int iMax = i0 + 15;
  // first chunk that can touch mask/lag region: j0 > i0 - 36
  const int jSplit = (i0 >= 36) ? ((((i0 - 36) >> 5) << 5) + 32) : 0;

  int j0 = 0;
  // ---------------- far path: no mask, constant lag bias ----------------
  for (; j0 < jSplit; j0 += 32) {
    v16h kb0 = load_b_f16(Kb + (long)j0 * DIM_HEAD,            DIM_HEAD);
    v16h kb1 = load_b_f16(Kb + (long)j0 * DIM_HEAD + 32,       DIM_HEAD);
    v16h kb2 = load_b_f16(Kb + (long)(j0 + 16) * DIM_HEAD,     DIM_HEAD);
    v16h kb3 = load_b_f16(Kb + (long)(j0 + 16) * DIM_HEAD + 32, DIM_HEAD);
    v8f s0 = {}, s1 = {};
    s0 = wmma_f16(qa0, kb0, s0);
    s0 = wmma_f16(qa1, kb1, s0);
    s1 = wmma_f16(qa0, kb2, s1);
    s1 = wmma_f16(qa1, kb3, s1);
    // hoist V fragments: loads overlap the softmax VALU below
    v16h vb0 = load_b_f16(Vb + (long)0  * T + j0, T);
    v16h vb1 = load_b_f16(Vb + (long)16 * T + j0, T);
    v16h vb2 = load_b_f16(Vb + (long)32 * T + j0, T);
    v16h vb3 = load_b_f16(Vb + (long)48 * T + j0, T);

    float p0[8], p1[8], f[8];
#pragma unroll
    for (int v = 0; v < 8; ++v) {
      p0[v] = s0[v] * scale + lp5;
      p1[v] = s1[v] * scale + lp5;
    }
#pragma unroll
    for (int v = 0; v < 8; ++v) {
      float rm = fmaxf(p0[v], p1[v]);
#pragma unroll
      for (int msk = 8; msk >= 1; msk >>= 1)
        rm = fmaxf(rm, __shfl_xor(rm, msk, 32));
      const float mn = fmaxf(m[v], rm);
      f[v] = __expf(m[v] - mn);
      m[v] = mn;
      p0[v] = __expf(p0[v] - mn);
      p1[v] = __expf(p1[v] - mn);
      float rs = p0[v] + p1[v];
#pragma unroll
      for (int msk = 8; msk >= 1; msk >>= 1)
        rs += __shfl_xor(rs, msk, 32);
      lsum[v] = lsum[v] * f[v] + rs;
    }
#pragma unroll
    for (int n = 0; n < 4; ++n)
#pragma unroll
      for (int v = 0; v < 8; ++v) o[n][v] *= f[v];
#pragma unroll
    for (int v = 0; v < 8; ++v) {
      lds[w][v + hi][col]      = (_Float16)p0[v];
      lds[w][v + hi][col + 16] = (_Float16)p1[v];
    }
    asm volatile("s_wait_dscnt 0x0" ::: "memory");
    const v16h pa = load_a_f16(&lds[w][0][0], 40);
    o[0] = wmma_f16(pa, vb0, o[0]);
    o[1] = wmma_f16(pa, vb1, o[1]);
    o[2] = wmma_f16(pa, vb2, o[2]);
    o[3] = wmma_f16(pa, vb3, o[3]);
  }

  // ------------- near path: causal mask + lag-select epilogue -------------
  for (; j0 <= iMax; j0 += 32) {
    v16h kb0 = load_b_f16(Kb + (long)j0 * DIM_HEAD,            DIM_HEAD);
    v16h kb1 = load_b_f16(Kb + (long)j0 * DIM_HEAD + 32,       DIM_HEAD);
    v16h kb2 = load_b_f16(Kb + (long)(j0 + 16) * DIM_HEAD,     DIM_HEAD);
    v16h kb3 = load_b_f16(Kb + (long)(j0 + 16) * DIM_HEAD + 32, DIM_HEAD);
    v8f s0 = {}, s1 = {};
    s0 = wmma_f16(qa0, kb0, s0);
    s0 = wmma_f16(qa1, kb1, s0);
    s1 = wmma_f16(qa0, kb2, s1);
    s1 = wmma_f16(qa1, kb3, s1);
    v16h vb0 = load_b_f16(Vb + (long)0  * T + j0, T);
    v16h vb1 = load_b_f16(Vb + (long)16 * T + j0, T);
    v16h vb2 = load_b_f16(Vb + (long)32 * T + j0, T);
    v16h vb3 = load_b_f16(Vb + (long)48 * T + j0, T);

    float p0[8], p1[8], f[8];
#pragma unroll
    for (int v = 0; v < 8; ++v) {
      const int i = i0 + v + hi;
      int dl = i - (j0 + col);
      p0[v] = (dl >= 0) ? (s0[v] * scale + lag_sel(lp, dl)) : -1e30f;
      dl = i - (j0 + 16 + col);
      p1[v] = (dl >= 0) ? (s1[v] * scale + lag_sel(lp, dl)) : -1e30f;
    }
#pragma unroll
    for (int v = 0; v < 8; ++v) {
      float rm = fmaxf(p0[v], p1[v]);
#pragma unroll
      for (int msk = 8; msk >= 1; msk >>= 1)
        rm = fmaxf(rm, __shfl_xor(rm, msk, 32));
      const float mn = fmaxf(m[v], rm);
      f[v] = __expf(m[v] - mn);
      m[v] = mn;
      p0[v] = __expf(p0[v] - mn);
      p1[v] = __expf(p1[v] - mn);
      float rs = p0[v] + p1[v];
#pragma unroll
      for (int msk = 8; msk >= 1; msk >>= 1)
        rs += __shfl_xor(rs, msk, 32);
      lsum[v] = lsum[v] * f[v] + rs;
    }
#pragma unroll
    for (int n = 0; n < 4; ++n)
#pragma unroll
      for (int v = 0; v < 8; ++v) o[n][v] *= f[v];
#pragma unroll
    for (int v = 0; v < 8; ++v) {
      lds[w][v + hi][col]      = (_Float16)p0[v];
      lds[w][v + hi][col + 16] = (_Float16)p1[v];
    }
    asm volatile("s_wait_dscnt 0x0" ::: "memory");
    const v16h pa = load_a_f16(&lds[w][0][0], 40);
    o[0] = wmma_f16(pa, vb0, o[0]);
    o[1] = wmma_f16(pa, vb1, o[1]);
    o[2] = wmma_f16(pa, vb2, o[2]);
    o[3] = wmma_f16(pa, vb3, o[3]);
  }

  // ---- normalize and store attention output [B*T, INNER] f16 ----
  float inv[8];
#pragma unroll
  for (int v = 0; v < 8; ++v) inv[v] = 1.f / lsum[v];
  _Float16* Aout = Ah + ((long)b * T + i0) * INNER + h * DIM_HEAD;
#pragma unroll
  for (int n = 0; n < 4; ++n)
#pragma unroll
    for (int v = 0; v < 8; ++v)
      Aout[(long)(v + hi) * INNER + n * 16 + col] = (_Float16)(o[n][v] * inv[v]);
}

// ---------------------------------------------------------------------------
// Kernel 4: output projection  out = Ah @ Wo + bo   (f32 epilogue),
// 32-row wave tile, double-buffered like proj_qkv.
// ---------------------------------------------------------------------------
__global__ void __launch_bounds__(128, 1)
proj_out_kernel(const _Float16* __restrict__ Ah,
                const _Float16* __restrict__ WoT,
                const float* __restrict__ bo,
                float* __restrict__ out) {
  const int w  = threadIdx.x >> 5;
  const int l  = threadIdx.x & 31;
  const int i0 = blockIdx.x * 128 + w * 32;
  const int n0 = blockIdx.y * 64;
  v8f acc0[4] = {}, acc1[4] = {};

  v16h a0 = load_a_f16(Ah + (long)i0 * INNER, INNER);
  v16h a1 = load_a_f16(Ah + (long)(i0 + 16) * INNER, INNER);
  v16h b0 = load_b_f16(WoT + (long)(n0 +  0) * INNER, INNER);
  v16h b1 = load_b_f16(WoT + (long)(n0 + 16) * INNER, INNER);
  v16h b2 = load_b_f16(WoT + (long)(n0 + 32) * INNER, INNER);
  v16h b3 = load_b_f16(WoT + (long)(n0 + 48) * INNER, INNER);

  for (int k0 = 32; k0 <= INNER - 32; k0 += 32) {
    v16h na0 = load_a_f16(Ah + (long)i0 * INNER + k0, INNER);
    v16h na1 = load_a_f16(Ah + (long)(i0 + 16) * INNER + k0, INNER);
    v16h nb0 = load_b_f16(WoT + (long)(n0 +  0) * INNER + k0, INNER);
    v16h nb1 = load_b_f16(WoT + (long)(n0 + 16) * INNER + k0, INNER);
    v16h nb2 = load_b_f16(WoT + (long)(n0 + 32) * INNER + k0, INNER);
    v16h nb3 = load_b_f16(WoT + (long)(n0 + 48) * INNER + k0, INNER);
    acc0[0] = wmma_f16(a0, b0, acc0[0]);
    acc0[1] = wmma_f16(a0, b1, acc0[1]);
    acc0[2] = wmma_f16(a0, b2, acc0[2]);
    acc0[3] = wmma_f16(a0, b3, acc0[3]);
    acc1[0] = wmma_f16(a1, b0, acc1[0]);
    acc1[1] = wmma_f16(a1, b1, acc1[1]);
    acc1[2] = wmma_f16(a1, b2, acc1[2]);
    acc1[3] = wmma_f16(a1, b3, acc1[3]);
    a0 = na0; a1 = na1; b0 = nb0; b1 = nb1; b2 = nb2; b3 = nb3;
  }
  acc0[0] = wmma_f16(a0, b0, acc0[0]);
  acc0[1] = wmma_f16(a0, b1, acc0[1]);
  acc0[2] = wmma_f16(a0, b2, acc0[2]);
  acc0[3] = wmma_f16(a0, b3, acc0[3]);
  acc1[0] = wmma_f16(a1, b0, acc1[0]);
  acc1[1] = wmma_f16(a1, b1, acc1[1]);
  acc1[2] = wmma_f16(a1, b2, acc1[2]);
  acc1[3] = wmma_f16(a1, b3, acc1[3]);

  const int hi = (l >> 4) * 8;
#pragma unroll
  for (int half = 0; half < 2; ++half) {
    const v8f* acc = half ? acc1 : acc0;
    const int ibase = i0 + half * 16;
#pragma unroll
    for (int n = 0; n < 4; ++n) {
      const int c = n0 + n * 16 + (l & 15);
      const float bias = bo[c];
#pragma unroll
      for (int v = 0; v < 8; ++v) {
        const int r = ibase + v + hi;
        out[(long)r * D + c] = acc[n][v] + bias;
      }
    }
  }
}

// ---------------------------------------------------------------------------
extern "C" void kernel_launch(void* const* d_in, const int* in_sizes, int n_in,
                              void* d_out, int out_size, void* d_ws, size_t ws_size,
                              hipStream_t stream) {
  const float* x    = (const float*)d_in[0];
  const float* Wq   = (const float*)d_in[1];
  const float* Wk   = (const float*)d_in[2];
  const float* Wv   = (const float*)d_in[3];
  const float* Wo   = (const float*)d_in[4];
  const float* bo   = (const float*)d_in[5];
  const float* lagw = (const float*)d_in[6];
  float* out = (float*)d_out;

  char* ws = (char*)d_ws;
  const size_t WSZ  = (size_t)D * INNER * sizeof(_Float16);        // 512 KB
  const size_t HSZ  = (size_t)BATCH * HEADS * T * DIM_HEAD * sizeof(_Float16); // 4 MB
  _Float16* WqT = (_Float16*)ws;              ws += WSZ;
  _Float16* WkT = (_Float16*)ws;              ws += WSZ;
  _Float16* WvT = (_Float16*)ws;              ws += WSZ;
  _Float16* WoT = (_Float16*)ws;              ws += WSZ;
  _Float16* Xh  = (_Float16*)ws;              ws += HSZ;   // [B*T, D] f16
  _Float16* Qh  = (_Float16*)ws;              ws += HSZ;
  _Float16* Kh  = (_Float16*)ws;              ws += HSZ;
  _Float16* Vt  = (_Float16*)ws;              ws += HSZ;   // [b,h,dh,T]
  _Float16* Ah  = (_Float16*)ws;              ws += HSZ;   // [B*T, INNER]

  // 0) x -> f16
  const int n4 = MROWS * D / 4;
  xcvt_kernel<<<(n4 + 255) / 256, 256, 0, stream>>>(x, Xh, n4);

  // 1) weight convert+transpose
  const int we = D * INNER;
  wtrans_kernel<<<(we + 255) / 256, 256, 0, stream>>>(Wq, WqT, D, INNER);
  wtrans_kernel<<<(we + 255) / 256, 256, 0, stream>>>(Wk, WkT, D, INNER);
  wtrans_kernel<<<(we + 255) / 256, 256, 0, stream>>>(Wv, WvT, D, INNER);
  wtrans_kernel<<<(we + 255) / 256, 256, 0, stream>>>(Wo, WoT, INNER, D);

  // 2) projections (WMMA, 128-row blocks)
  dim3 gproj(MROWS / 128, INNER / 64);
  proj_qkv_kernel<<<gproj, 128, 0, stream>>>(Xh, WqT, Qh, 0);
  proj_qkv_kernel<<<gproj, 128, 0, stream>>>(Xh, WkT, Kh, 0);
  proj_qkv_kernel<<<gproj, 128, 0, stream>>>(Xh, WvT, Vt, 1);

  // 3) flash attention (WMMA)
  dim3 gfa(T / 64, BATCH * HEADS);
  flash_kernel<<<gfa, 128, 0, stream>>>(Qh, Kh, Vt, lagw, Ah);

  // 4) output projection (WMMA)
  proj_out_kernel<<<gproj, 128, 0, stream>>>(Ah, WoT, bo, out);
}
